// ModuleWithExtendedAttention_8134668058666
// MI455X (gfx1250) — matrix operand order repaired and checked by reading, not compiled, not run on hardware
//
#include <hip/hip_runtime.h>
#include <hip/hip_bf16.h>

typedef _Float16 half_t;
typedef __attribute__((ext_vector_type(16))) _Float16 v16h;
typedef __attribute__((ext_vector_type(8)))  _Float16 v8h;
typedef __attribute__((ext_vector_type(8)))  float    v8f;
typedef __attribute__((ext_vector_type(4)))  unsigned int u32x4;
typedef __attribute__((ext_vector_type(8)))  int      i32x8;
typedef __attribute__((ext_vector_type(4)))  int      i32x4;

// ---- problem constants ----
#define DD    1024
#define TEXTN 226
#define STN   1248
#define SN    1474      // TEXT + ST
#define SP    1488      // SN padded to 16  (93 tiles)
#define SKC   2722      // SN + STN (cond keys)
#define SKCP  2752      // padded to 32 (86 blocks)
#define SKUP  1504      // SN padded to 32 (47 blocks)
#define NEGB  (-1000000000.0f)

// Tensor Data Mover availability (probe-verified builtin on both toolchains,
// with different arity; guard keeps compile safe either way).
#if defined(__has_builtin)
#  if __has_builtin(__builtin_amdgcn_tensor_load_to_lds)
#    define USE_TDM 1
#  endif
#endif
#ifndef USE_TDM
#  define USE_TDM 0
#endif

// ---------------- helpers ----------------
__device__ inline v16h load_a16(const half_t* rowp, int hs) {
  // f16 A 16x32 layout: lane m=lane&15, hs=lane>>4
  // elems 0..15 -> K = hs*16 + i (contiguous, matches B)
  v8h x0 = *(const v8h*)(rowp + hs * 16);
  v8h x1 = *(const v8h*)(rowp + hs * 16 + 8);
  v16h a;
#pragma unroll
  for (int i = 0; i < 8; i++) { a[i] = x0[i]; a[i + 8] = x1[i]; }
  return a;
}

__device__ inline v16h load_b16(const half_t* p) {
  // B 32x16: lane n=lane&15 (col), kh=lane>>4; 16 contiguous K halfs
  v8h x0 = *(const v8h*)(p);
  v8h x1 = *(const v8h*)(p + 8);
  v16h b;
#pragma unroll
  for (int i = 0; i < 8; i++) { b[i] = x0[i]; b[i + 8] = x1[i]; }
  return b;
}

__device__ inline v8f wmma_f16(v16h a, v16h b, v8f c) {
  return __builtin_amdgcn_wmma_f32_16x16x32_f16(false, a, false, b, (short)0, c,
                                                false, false);
}

__device__ inline float red_max16(float v) {
  v = fmaxf(v, __shfl_xor(v, 1, 32));
  v = fmaxf(v, __shfl_xor(v, 2, 32));
  v = fmaxf(v, __shfl_xor(v, 4, 32));
  v = fmaxf(v, __shfl_xor(v, 8, 32));
  return v;
}
__device__ inline float red_sum16(float v) {
  v += __shfl_xor(v, 1, 32);
  v += __shfl_xor(v, 2, 32);
  v += __shfl_xor(v, 4, 32);
  v += __shfl_xor(v, 8, 32);
  return v;
}

#if USE_TDM
// Issue a 2D TDM tile load global->LDS.  All args wave-uniform.
// rows x cols elements of 2 bytes; row stride = stride_elems.
__device__ inline void tdm_load_2d(unsigned int lds_off, const void* gptr,
                                   unsigned int rows, unsigned int cols,
                                   unsigned int stride_elems) {
  unsigned long long ga = (unsigned long long)gptr;
  u32x4 g0;
  g0[0] = 1u;                                        // count=1 (valid), user D#
  g0[1] = lds_off;                                   // lds_addr (bytes)
  g0[2] = (unsigned int)(ga & 0xFFFFFFFFu);          // global_addr[31:0]
  g0[3] = (unsigned int)((ga >> 32) & 0x1FFFFFFu)    // global_addr[56:32]
          | (2u << 30);                              // type = 2 ("image")
  i32x8 g1;
  g1[0] = (int)(1u << 16);        // wg_mask=0, data_size=1 (2 bytes)
  g1[1] = (int)(cols << 16);      // tensor_dim0[15:0]  (desc bits 63:48)
  g1[2] = (int)(rows << 16);      // tensor_dim1[15:0]  (desc bits 95:80)
  g1[3] = (int)(cols << 16);      // tile_dim0          (desc bits 127:112)
  g1[4] = (int)rows;              // tile_dim1          (desc bits 143:128)
  g1[5] = (int)stride_elems;      // tensor_dim0_stride[31:0]
  g1[6] = 0;
  g1[7] = 0;
  i32x4 z4 = {0, 0, 0, 0};
#if __clang_major__ >= 23
  i32x8 z8 = {0, 0, 0, 0, 0, 0, 0, 0};
  __builtin_amdgcn_tensor_load_to_lds(g0, g1, z4, z4, z8, 0);
#else
  __builtin_amdgcn_tensor_load_to_lds(g0, g1, z4, z4, 0);
#endif
}

__device__ inline void wait_tensor_le2() {
#if __has_builtin(__builtin_amdgcn_s_wait_tensorcnt)
  __builtin_amdgcn_s_wait_tensorcnt(2);
#else
  asm volatile("s_wait_tensorcnt 0x2" ::: "memory");
#endif
}
__device__ inline void wait_tensor_le0() {
#if __has_builtin(__builtin_amdgcn_s_wait_tensorcnt)
  __builtin_amdgcn_s_wait_tensorcnt(0);
#else
  asm volatile("s_wait_tensorcnt 0x0" ::: "memory");
#endif
}
#endif  // USE_TDM

// ---------------- prep kernels ----------------
__global__ __launch_bounds__(256) void prep_x_kernel(
    const float* __restrict__ hidden, const float* __restrict__ enc,
    half_t* __restrict__ xh) {
  size_t idx = (size_t)blockIdx.x * 256 + threadIdx.x;
  size_t total = (size_t)2 * SP * DD;
  if (idx >= total) return;
  int b = (int)(idx / ((size_t)SP * DD));
  size_t rem = idx - (size_t)b * SP * DD;
  int row = (int)(rem / DD);
  int d = (int)(rem % DD);
  float v = 0.0f;
  if (row < TEXTN)      v = enc[((size_t)b * TEXTN + row) * DD + d];
  else if (row < SN)    v = hidden[((size_t)b * STN + (row - TEXTN)) * DD + d];
  xh[idx] = (half_t)v;
}

__global__ __launch_bounds__(256) void prep_w_kernel(
    const float* __restrict__ wq, const float* __restrict__ wk,
    const float* __restrict__ wv, const float* __restrict__ wo,
    half_t* __restrict__ wqt, half_t* __restrict__ wkt,
    half_t* __restrict__ wvt, half_t* __restrict__ wot) {
  size_t idx = (size_t)blockIdx.x * 256 + threadIdx.x;
  if (idx >= (size_t)DD * DD) return;
  int k = (int)(idx / DD);
  int n = (int)(idx % DD);
  size_t dst = (size_t)n * DD + k;   // N-major (transposed)
  wqt[dst] = (half_t)wq[idx];
  wkt[dst] = (half_t)wk[idx];
  wvt[dst] = (half_t)wv[idx];
  wot[dst] = (half_t)wo[idx];
}

__global__ __launch_bounds__(256) void prep_bias_kernel(
    const int* __restrict__ mask, float* __restrict__ biasC,
    float* __restrict__ biasU) {
  int idx = blockIdx.x * 256 + threadIdx.x;
  if (idx < SKCP) {
    float v;
    if (idx < SN)       v = 0.0f;
    else if (idx < SKC) v = (mask[idx - SN] > 0) ? 0.0f : NEGB;
    else                v = NEGB;
    biasC[idx] = v;
  }
  int j = idx - SKCP;
  if (j >= 0 && j < SKUP) biasU[j] = (j < SN) ? 0.0f : NEGB;
}

// ---------------- QKV GEMM (WMMA) ----------------
__global__ __launch_bounds__(32) void gemm_qkv_kernel(
    const half_t* __restrict__ xh, const half_t* __restrict__ wqt,
    const half_t* __restrict__ wkt, const half_t* __restrict__ wvt,
    const float* __restrict__ bq, const float* __restrict__ bk,
    const float* __restrict__ bv, half_t* __restrict__ qh,
    half_t* __restrict__ kh, half_t* __restrict__ vh) {
  const int lane = threadIdx.x;
  const int m = lane & 15, hs = lane >> 4;
  const int n = lane & 15, khh = lane >> 4;
  const int nst = blockIdx.x;       // 16 tiles of 64 cols
  const int mt = blockIdx.y;        // 93 tiles of 16 rows
  const int b = blockIdx.z;
  const size_t boff = (size_t)b * SP * DD;
  const half_t* arow = xh + boff + (size_t)(mt * 16 + m) * DD;
  const int col0 = nst * 64;

  v8f accq[4] = {}, acck[4] = {}, accv[4] = {};
  for (int k0 = 0; k0 < DD; k0 += 32) {
    v16h a = load_a16(arow + k0, hs);
#pragma unroll
    for (int c = 0; c < 4; c++) {
      size_t wrow = (size_t)(col0 + 16 * c + n) * DD + k0 + khh * 16;
      accq[c] = wmma_f16(a, load_b16(wqt + wrow), accq[c]);
      acck[c] = wmma_f16(a, load_b16(wkt + wrow), acck[c]);
      accv[c] = wmma_f16(a, load_b16(wvt + wrow), accv[c]);
    }
  }
#pragma unroll
  for (int c = 0; c < 4; c++) {
    int col = col0 + 16 * c + n;
    float bqv = bq[col], bkv = bk[col], bvv = bv[col];
#pragma unroll
    for (int r = 0; r < 8; r++) {
      size_t idx = boff + (size_t)(mt * 16 + r + 8 * hs) * DD + col;
      qh[idx] = (half_t)(accq[c][r] + bqv);
      kh[idx] = (half_t)(acck[c][r] + bkv);
      vh[idx] = (half_t)(accv[c][r] + bvv);
    }
  }
}

// ---------------- K/V packing ----------------
__global__ __launch_bounds__(256) void pack_cond_kernel(
    const half_t* __restrict__ kh, const half_t* __restrict__ vh,
    half_t* __restrict__ Kc, half_t* __restrict__ VcT) {
  size_t idx = (size_t)blockIdx.x * 256 + threadIdx.x;
  if (idx >= (size_t)SKCP * DD) return;
  int j = (int)(idx / DD);
  int d = (int)(idx % DD);
  half_t kv = (half_t)0.0f, vv = (half_t)0.0f;
  if (j < SN) {
    size_t src = (size_t)j * DD + d;                        // batch 0
    kv = kh[src]; vv = vh[src];
  } else if (j < SKC) {
    size_t src = ((size_t)SP + TEXTN + (j - SN)) * DD + d;  // batch 1, ST part
    kv = kh[src]; vv = vh[src];
  }
  Kc[idx] = kv;
  VcT[(size_t)d * SKCP + j] = vv;   // dim-major (key-contiguous)
}

__global__ __launch_bounds__(256) void pack_uncond_kernel(
    const half_t* __restrict__ kh, const half_t* __restrict__ vh,
    half_t* __restrict__ Ku, half_t* __restrict__ VuT) {
  size_t idx = (size_t)blockIdx.x * 256 + threadIdx.x;
  if (idx >= (size_t)SKUP * DD) return;
  int j = (int)(idx / DD);
  int d = (int)(idx % DD);
  half_t kv = (half_t)0.0f, vv = (half_t)0.0f;
  if (j < SN) {
    size_t src = ((size_t)SP + j) * DD + d;                 // batch 1
    kv = kh[src]; vv = vh[src];
  }
  Ku[idx] = kv;
  VuT[(size_t)d * SKUP + j] = vv;
}

// ---------------- flash attention (WMMA + TDM double buffering) ----------
__global__ __launch_bounds__(32) void attn_kernel(
    const half_t* __restrict__ qh, const half_t* __restrict__ Kc,
    const half_t* __restrict__ VcT, const float* __restrict__ biasC,
    const half_t* __restrict__ Ku, const half_t* __restrict__ VuT,
    const float* __restrict__ biasU, half_t* __restrict__ attn) {
  const int lane = threadIdx.x;
  const int m = lane & 15, hs = lane >> 4;
  const int n = lane & 15, khh = lane >> 4;
  const int qtile = blockIdx.x;     // 93
  const int head = blockIdx.y;      // 16
  const int z = blockIdx.z;         // 0=cond, 1=uncond
  const int hoff = head * 64;

  const half_t* K; const half_t* Vt; const float* bias;
  int SK, ldv; size_t qoff;
  if (z == 0) { K = Kc; Vt = VcT; bias = biasC; SK = SKCP; ldv = SKCP; qoff = 0; }
  else        { K = Ku; Vt = VuT; bias = biasU; SK = SKUP; ldv = SKUP;
                qoff = (size_t)SP * DD; }

  const int row0 = qtile * 16;
  const half_t* qrow = qh + qoff + (size_t)(row0 + m) * DD + hoff;
  v16h a0 = load_a16(qrow, hs);        // q dims 0..31
  v16h a1 = load_a16(qrow + 32, hs);   // q dims 32..63

  float mi[8], li[8];
  v8f o0 = {}, o1 = {}, o2 = {}, o3 = {};
#pragma unroll
  for (int r = 0; r < 8; r++) { mi[r] = -3.0e38f; li[r] = 0.0f; }

  __shared__ half_t ldsp[16 * 32];     // P tile (C-layout -> A-layout)
#if USE_TDM
  __shared__ half_t kbuf[2][32 * 64];  // K block: 32 keys x 64 dims
  __shared__ half_t vbuf[2][64 * 32];  // V block: 64 dims x 32 keys
  // prologue: DMA block 0 into buffer 0
  tdm_load_2d((unsigned int)(size_t)&kbuf[0][0], K + /*kb=*/0 * (size_t)DD + hoff,
              /*rows=*/32, /*cols=*/64, /*stride=*/DD);
  tdm_load_2d((unsigned int)(size_t)&vbuf[0][0], Vt + (size_t)hoff * ldv + /*kb=*/0,
              /*rows=*/64, /*cols=*/32, /*stride=*/(unsigned int)ldv);
#endif
  const float scale = 0.125f;          // 1/sqrt(64)

  for (int kb = 0; kb < SK; kb += 32) {
#if USE_TDM
    const int buf = (kb >> 5) & 1;
    const int kn = kb + 32;
    if (kn < SK) {   // uniform branch: prefetch next block, then wait current
      tdm_load_2d((unsigned int)(size_t)&kbuf[buf ^ 1][0],
                  K + (size_t)kn * DD + hoff, 32, 64, DD);
      tdm_load_2d((unsigned int)(size_t)&vbuf[buf ^ 1][0],
                  Vt + (size_t)hoff * ldv + kn, 64, 32, (unsigned int)ldv);
      wait_tensor_le2();   // in-order: <=2 outstanding => current buffer ready
    } else {
      wait_tensor_le0();
    }
    __syncthreads();
#endif

    // ---- scores: two 16-key subtiles, K-dim 64 = 2 WMMA chunks each ----
    v8f s0 = {}, s1 = {};
    {
#if USE_TDM
      const half_t* kp0 = &kbuf[buf][n * 64 + khh * 16];
      const half_t* kp1 = &kbuf[buf][(16 + n) * 64 + khh * 16];
#else
      const half_t* kp0 = K + (size_t)(kb + n) * DD + hoff + khh * 16;
      const half_t* kp1 = K + (size_t)(kb + 16 + n) * DD + hoff + khh * 16;
#endif
      s0 = wmma_f16(a0, load_b16(kp0), s0);
      s0 = wmma_f16(a1, load_b16(kp0 + 32), s0);
      s1 = wmma_f16(a0, load_b16(kp1), s1);
      s1 = wmma_f16(a1, load_b16(kp1 + 32), s1);
    }
    const float b0v = bias[kb + n];
    const float b1v = bias[kb + 16 + n];

    // ---- online softmax (rows = r + 8*hs, cols across 16 lanes) ----
    float p0[8], p1[8], alpha[8];
#pragma unroll
    for (int r = 0; r < 8; r++) {
      float x0 = s0[r] * scale + b0v;
      float x1 = s1[r] * scale + b1v;
      float t = red_max16(fmaxf(x0, x1));
      float mn = fmaxf(mi[r], t);
      alpha[r] = __expf(mi[r] - mn);
      p0[r] = __expf(x0 - mn);
      p1[r] = __expf(x1 - mn);
      float rs = red_sum16(p0[r] + p1[r]);
      li[r] = li[r] * alpha[r] + rs;
      mi[r] = mn;
    }
#pragma unroll
    for (int r = 0; r < 8; r++) {
      o0[r] = o0[r] * alpha[r];
      o1[r] = o1[r] * alpha[r];
      o2[r] = o2[r] * alpha[r];
      o3[r] = o3[r] * alpha[r];
    }

    // ---- stage P through LDS: C-layout -> row-major [16][32] ----
#pragma unroll
    for (int r = 0; r < 8; r++) {
      int mm = r + 8 * hs;
      ldsp[mm * 32 + n] = (half_t)p0[r];
      ldsp[mm * 32 + 16 + n] = (half_t)p1[r];
    }
    __syncthreads();
    v16h pa = load_a16(&ldsp[m * 32], hs);   // A-layout 16x32
    __syncthreads();

    // ---- PV: B = V[32 keys x 16 dims] ----
#if USE_TDM
    const half_t* vb0 = &vbuf[buf][n * 32 + khh * 16];
    o0 = wmma_f16(pa, load_b16(vb0), o0);
    o1 = wmma_f16(pa, load_b16(vb0 + 16 * 32), o1);
    o2 = wmma_f16(pa, load_b16(vb0 + 32 * 32), o2);
    o3 = wmma_f16(pa, load_b16(vb0 + 48 * 32), o3);
#else
    const half_t* vb = Vt + (size_t)(hoff + n) * ldv + kb + khh * 16;
    o0 = wmma_f16(pa, load_b16(vb), o0);
    o1 = wmma_f16(pa, load_b16(vb + (size_t)16 * ldv), o1);
    o2 = wmma_f16(pa, load_b16(vb + (size_t)32 * ldv), o2);
    o3 = wmma_f16(pa, load_b16(vb + (size_t)48 * ldv), o3);
#endif
  }

  // ---- normalize + store f16 ----
#pragma unroll
  for (int r = 0; r < 8; r++) {
    int row = row0 + r + 8 * hs;
    if (row < SN) {
      float inv = 1.0f / li[r];
      size_t base = qoff + (size_t)row * DD + hoff + n;
      attn[base]      = (half_t)(o0[r] * inv);
      attn[base + 16] = (half_t)(o1[r] * inv);
      attn[base + 32] = (half_t)(o2[r] * inv);
      attn[base + 48] = (half_t)(o3[r] * inv);
    }
  }
}

// ---------------- output GEMM (WMMA, writes d_out with split order) -------
__global__ __launch_bounds__(32) void gemm_out_kernel(
    const half_t* __restrict__ ah, const half_t* __restrict__ wot,
    const float* __restrict__ bo, float* __restrict__ out) {
  const int lane = threadIdx.x;
  const int m = lane & 15, hs = lane >> 4;
  const int n = lane & 15, khh = lane >> 4;
  const int nst = blockIdx.x;       // 16
  const int mt = blockIdx.y;        // 93
  const int b = blockIdx.z;
  const size_t boff = (size_t)b * SP * DD;
  const half_t* arow = ah + boff + (size_t)(mt * 16 + m) * DD;
  const int col0 = nst * 64;

  v8f acc[4] = {};
  for (int k0 = 0; k0 < DD; k0 += 32) {
    v16h a = load_a16(arow + k0, hs);
#pragma unroll
    for (int c = 0; c < 4; c++) {
      size_t wrow = (size_t)(col0 + 16 * c + n) * DD + k0 + khh * 16;
      acc[c] = wmma_f16(a, load_b16(wot + wrow), acc[c]);
    }
  }
#pragma unroll
  for (int c = 0; c < 4; c++) {
    int col = col0 + 16 * c + n;
    float bov = bo[col];
#pragma unroll
    for (int r = 0; r < 8; r++) {
      int s = mt * 16 + r + 8 * hs;
      if (s < SN) {
        size_t dst;
        if (s >= TEXTN)   // first tuple element: out[:, TEXT:, :]
          dst = ((size_t)b * STN + (s - TEXTN)) * DD + col;
        else              // second tuple element after 2*ST*D floats
          dst = (size_t)2 * STN * DD + ((size_t)b * TEXTN + s) * DD + col;
        out[dst] = acc[c][r] + bov;
      }
    }
  }
}

// ---------------- host launcher ----------------
extern "C" void kernel_launch(void* const* d_in, const int* in_sizes, int n_in,
                              void* d_out, int out_size, void* d_ws,
                              size_t ws_size, hipStream_t stream) {
  const float* hidden = (const float*)d_in[0];
  const float* enc    = (const float*)d_in[1];
  const int*   mask   = (const int*)d_in[2];
  const float* wq = (const float*)d_in[3]; const float* bq = (const float*)d_in[4];
  const float* wk = (const float*)d_in[5]; const float* bk = (const float*)d_in[6];
  const float* wv = (const float*)d_in[7]; const float* bv = (const float*)d_in[8];
  const float* wo = (const float*)d_in[9]; const float* bo = (const float*)d_in[10];
  float* out = (float*)d_out;

  // workspace carve-up (256B aligned)
  char* ws = (char*)d_ws;
  size_t off = 0;
  auto carve = [&](size_t bytes) -> char* {
    char* p = ws + off;
    off = (off + bytes + 255) & ~(size_t)255;
    return p;
  };
  half_t* xh   = (half_t*)carve((size_t)2 * SP * DD * 2);
  half_t* wqt  = (half_t*)carve((size_t)DD * DD * 2);
  half_t* wkt  = (half_t*)carve((size_t)DD * DD * 2);
  half_t* wvt  = (half_t*)carve((size_t)DD * DD * 2);
  half_t* wot  = (half_t*)carve((size_t)DD * DD * 2);
  half_t* qh   = (half_t*)carve((size_t)2 * SP * DD * 2);
  half_t* kh   = (half_t*)carve((size_t)2 * SP * DD * 2);
  half_t* vh   = (half_t*)carve((size_t)2 * SP * DD * 2);
  half_t* Kc   = (half_t*)carve((size_t)SKCP * DD * 2);
  half_t* VcT  = (half_t*)carve((size_t)SKCP * DD * 2);
  half_t* Ku   = (half_t*)carve((size_t)SKUP * DD * 2);
  half_t* VuT  = (half_t*)carve((size_t)SKUP * DD * 2);
  float*  biasC = (float*)carve((size_t)SKCP * 4);
  float*  biasU = (float*)carve((size_t)SKUP * 4);
  half_t* attn = (half_t*)carve((size_t)2 * SP * DD * 2);
  (void)ws_size; (void)in_sizes; (void)n_in; (void)out_size;

  // 1) prep
  {
    size_t N = (size_t)2 * SP * DD;
    prep_x_kernel<<<(unsigned)((N + 255) / 256), 256, 0, stream>>>(hidden, enc, xh);
  }
  {
    size_t N = (size_t)DD * DD;
    prep_w_kernel<<<(unsigned)((N + 255) / 256), 256, 0, stream>>>(
        wq, wk, wv, wo, wqt, wkt, wvt, wot);
  }
  prep_bias_kernel<<<(SKCP + SKUP + 255) / 256, 256, 0, stream>>>(mask, biasC, biasU);

  // 2) QKV projections
  gemm_qkv_kernel<<<dim3(16, SP / 16, 2), 32, 0, stream>>>(
      xh, wqt, wkt, wvt, bq, bk, bv, qh, kh, vh);

  // 3) pack extended K/V
  {
    size_t N = (size_t)SKCP * DD;
    pack_cond_kernel<<<(unsigned)((N + 255) / 256), 256, 0, stream>>>(kh, vh, Kc, VcT);
  }
  {
    size_t N = (size_t)SKUP * DD;
    pack_uncond_kernel<<<(unsigned)((N + 255) / 256), 256, 0, stream>>>(kh, vh, Ku, VuT);
  }

  // 4) attention (cond + uncond via grid.z)
  attn_kernel<<<dim3(SP / 16, 16, 2), 32, 0, stream>>>(
      qh, Kc, VcT, biasC, Ku, VuT, biasU, attn);

  // 5) output projection -> d_out (split tuple ordering)
  gemm_out_kernel<<<dim3(16, SP / 16, 2), 32, 0, stream>>>(attn, wot, bo, out);
}
